// MaskEmbdMultiMPNV2_82858509074791
// MI455X (gfx1250) — matrix operand order repaired and checked by reading, not compiled, not run on hardware
//
#include <hip/hip_runtime.h>

// ---------------------------------------------------------------------------
// CDNA5 (gfx1250) GNN forward: MaskEmbdMultiMPNV2
// All dense math runs through v_wmma_f32_16x16x32_bf16 (wave32 WMMA).
// Weights are pre-transposed to [N][K] bf16 so B fragments load straight from
// global memory (2x global_load_b128 per fragment, L2-resident); A tiles are
// staged through LDS with aligned b128 traffic only.
// ---------------------------------------------------------------------------

typedef __bf16 bf16;
typedef __attribute__((ext_vector_type(16))) __bf16 v16bf;
typedef __attribute__((ext_vector_type(8)))  __bf16 v8bf;   // 16 bytes
typedef __attribute__((ext_vector_type(8)))  float  v8f;

#define DEV __device__ __forceinline__

DEV float silu_f(float x) { return x / (1.0f + __expf(-x)); }

DEV v8f wmma_bf16(v16bf a, v16bf b, v8f c) {
  return __builtin_amdgcn_wmma_f32_16x16x32_bf16(false, a, false, b, (short)0, c,
                                                 false, false);
}

// A fragment from a row-major [M][stride] LDS tile (stride % 8 == 0 so each
// half is a 16B-aligned ds_load_b128).  lane 0-15: M=lane, K-base 0; lane
// 16-31: M=lane-16, K-base 8.  elems 0..7 -> K=kb..kb+7, 8..15 -> K=16+kb...
DEV v16bf load_a_frag_rm(const bf16* lds, int row_base, int col_base, int stride,
                         int lane) {
  int m = row_base + (lane & 15);
  int kb = col_base + ((lane >> 4) << 3);
  const bf16* p = lds + m * stride + kb;
  union { v16bf v; v8bf h[2]; } u;
  u.h[0] = *(const v8bf*)p;
  u.h[1] = *(const v8bf*)(p + 16);
  return u.v;
}

// B fragment loaded DIRECTLY from a transposed global weight Bt[N][K]:
// lane 0-15: N=n_base+lane, K=kk..kk+7,kk+16..kk+23; lane 16-31: N=n_base+lane-16,
// K=kk+8..kk+15,kk+24..kk+31 -> 2x 16B chunks per lane (2x global_load_b128).
DEV v16bf load_b_frag_g(const bf16* __restrict__ Bt, int ldk, int n_base, int kk,
                        int lane) {
  const bf16* p = Bt + (long)(n_base + (lane & 15)) * ldk + kk + ((lane >> 4) << 3);
  union { v16bf v; v8bf h[2]; } u;
  u.h[0] = *(const v8bf*)p;
  u.h[1] = *(const v8bf*)(p + 16);
  return u.v;
}

// ---------------------------------------------------------------------------
// Generic bf16 GEMM:  Out[M,N] = act(A[M,K] @ Bt^T + bias) + Cin
// A row-major [M,K]; Bt transposed weights [N][K].
// 256 threads = 8 waves; block tile 32(M) x 128(N); K step 32.
// M%32==0, N%128==0, K%32==0 hold for every call site here.
// ---------------------------------------------------------------------------
template <int ACT>
__global__ __launch_bounds__(256) void gemm_kernel(
    const bf16* __restrict__ A, const bf16* __restrict__ Bt, const float* bias,
    const float* Cin, float* Out, bf16* OutBf, int M, int N, int K) {
  __shared__ __align__(16) bf16 sA[32][40];  // row-major A tile
  const int tid = threadIdx.x;
  const int lane = tid & 31;
  const int wave = tid >> 5;
  const int mw = wave & 1;   // 0..1 -> 16 rows each
  const int nw = wave >> 1;  // 0..3 -> 32 cols each
  const long row0 = (long)blockIdx.x * 32;
  const int col0 = blockIdx.y * 128;

  v8f acc[2] = {};
  for (int kk = 0; kk < K; kk += 32) {
    if (tid < 128) {  // A tile: one aligned 16B chunk per thread
      int r = tid >> 2, c8 = (tid & 3) << 3;
      *(v8bf*)&sA[r][c8] = *(const v8bf*)(A + (row0 + r) * (long)K + kk + c8);
    }
    if (kk + 32 < K)  // next weight slice toward L2 (global_prefetch_b8)
      __builtin_prefetch((const void*)(Bt + (long)(col0 + (tid & 127)) * K + kk + 32),
                         0, 1);
    __syncthreads();
    v16bf a = load_a_frag_rm(&sA[0][0], mw * 16, 0, 40, lane);
#pragma unroll
    for (int s = 0; s < 2; ++s) {
      v16bf b = load_b_frag_g(Bt, K, col0 + nw * 32 + s * 16, kk, lane);
      acc[s] = wmma_bf16(a, b, acc[s]);
    }
    __syncthreads();
  }
#pragma unroll
  for (int s = 0; s < 2; ++s) {
#pragma unroll
    for (int j = 0; j < 8; ++j) {
      long m = row0 + mw * 16 + ((lane >> 4) << 3) + j;
      long n = col0 + nw * 32 + s * 16 + (lane & 15);
      float v = acc[s][j];
      if (bias) v += bias[n];
      if (ACT) v = silu_f(v);
      if (Cin) v += Cin[m * N + n];
      if (Out) Out[m * N + n] = v;
      if (OutBf) OutBf[m * N + n] = (bf16)v;
    }
  }
}

// ---------------------------------------------------------------------------
// Fused edge message MLP:
//   for 32 edges: A = [x[dst] | x[src] | e]  (K1 = 2*NF+128)
//   hidden = silu(A@W1 + b1)   (32x256, kept in LDS as bf16)
//   msg    = hidden@W2 + b2    (32x128)
//   atomicAdd(out[dst], msg)   (segment-sum at target)
// Edge e < E uses (src=ei[0][e], dst=ei[1][e]); e >= E the reverse.
// w1t: [256][K1] transposed; w2t: [128][256] transposed.
// ---------------------------------------------------------------------------
template <int NF>
__global__ __launch_bounds__(256) void edge_mp_kernel(
    const int* __restrict__ ei, int E, const bf16* __restrict__ xbf,
    const bf16* __restrict__ ebf, const bf16* __restrict__ w1t,
    const float* __restrict__ b1, const bf16* __restrict__ w2t,
    const float* __restrict__ b2, float* __restrict__ out) {
  constexpr int K1 = 2 * NF + 128;
  __shared__ __align__(16) bf16 sA[32][40];   // gathered A tile (row-major)
  __shared__ __align__(16) bf16 sH[32][264];  // hidden (row-major, 16B rows)
  __shared__ int ssrc[32], sdst[32], seid[32];

  const int tid = threadIdx.x;
  const int lane = tid & 31;
  const int wave = tid >> 5;
  const int mw = wave & 1;   // 16 edges each
  const int nw = wave >> 1;  // layer1: 64 cols each ; layer2: 32 cols each
  const int ebase = blockIdx.x * 32;

  if (tid < 32) {
    int e = ebase + tid;
    int src, dst, eidx;
    if (e < E) { eidx = e;     src = ei[e];        dst = ei[E + e]; }
    else       { eidx = e - E; src = ei[E + eidx]; dst = ei[eidx]; }
    ssrc[tid] = src; sdst[tid] = dst; seid[tid] = eidx;
  }
  __syncthreads();

  // ---- layer 1: 32 x 256 = A(32xK1) @ W1(K1x256) ----
  v8f acc[4] = {};
  for (int kk = 0; kk < K1; kk += 32) {
    // gathered A tile: 8-wide chunks never straddle the NF/2NF boundaries
    if (tid < 128) {
      int r = tid >> 2, c8 = (tid & 3) << 3;
      int gc = kk + c8;
      const bf16* srcp;
      if (gc < NF)          srcp = xbf + (long)sdst[r] * NF + gc;
      else if (gc < 2 * NF) srcp = xbf + (long)ssrc[r] * NF + (gc - NF);
      else                  srcp = ebf + (long)seid[r] * 128 + (gc - 2 * NF);
      *(v8bf*)&sA[r][c8] = *(const v8bf*)srcp;
    }
    if (kk + 32 < K1)
      __builtin_prefetch((const void*)(w1t + (long)tid * K1 + kk + 32), 0, 1);
    __syncthreads();
    v16bf a = load_a_frag_rm(&sA[0][0], mw * 16, 0, 40, lane);
#pragma unroll
    for (int s = 0; s < 4; ++s) {
      v16bf b = load_b_frag_g(w1t, K1, nw * 64 + s * 16, kk, lane);
      acc[s] = wmma_bf16(a, b, acc[s]);
    }
    __syncthreads();
  }
  // bias + SiLU -> bf16 hidden in LDS
#pragma unroll
  for (int s = 0; s < 4; ++s) {
    int nb = nw * 64 + s * 16 + (lane & 15);
    int mb = mw * 16 + ((lane >> 4) << 3);
    float bia = b1[nb];
#pragma unroll
    for (int j = 0; j < 8; ++j) sH[mb + j][nb] = (bf16)silu_f(acc[s][j] + bia);
  }
  __syncthreads();

  // ---- layer 2: 32 x 128 = hidden(32x256) @ W2(256x128) ----
  v8f acc2[2] = {};
  for (int kk = 0; kk < 256; kk += 32) {
    v16bf a = load_a_frag_rm(&sH[0][0], mw * 16, kk, 264, lane);
#pragma unroll
    for (int s = 0; s < 2; ++s) {
      v16bf b = load_b_frag_g(w2t, 256, nw * 32 + s * 16, kk, lane);
      acc2[s] = wmma_bf16(a, b, acc2[s]);
    }
  }
  // scatter-add to target nodes
#pragma unroll
  for (int s = 0; s < 2; ++s) {
    int nb = nw * 32 + s * 16 + (lane & 15);
    int mb = mw * 16 + ((lane >> 4) << 3);
    float bia = b2[nb];
#pragma unroll
    for (int j = 0; j < 8; ++j)
      atomicAdd(&out[(long)sdst[mb + j] * 128 + nb], acc2[s][j] + bia);
  }
}

// ---------------------------------------------------------------------------
// Small elementwise / helper kernels
// ---------------------------------------------------------------------------
__global__ void convert_kernel(const float* __restrict__ in, bf16* __restrict__ out,
                               long n) {
  long i = (long)blockIdx.x * 256 + threadIdx.x;
  if (i < n) out[i] = (bf16)in[i];
}

// f32 [K,N] row-major  ->  bf16 [N,K] (transposed weight for B fragments)
__global__ void convert_tr_kernel(const float* __restrict__ in,
                                  bf16* __restrict__ out, int K, int N) {
  long i = (long)blockIdx.x * 256 + threadIdx.x;
  if (i >= (long)K * N) return;
  int k = (int)(i / N), n = (int)(i % N);
  out[(long)n * K + k] = (bf16)in[i];
}

__global__ void ew_kernel(const float* __restrict__ in, float* outf, bf16* outbf,
                          long n, int act) {
  long i = (long)blockIdx.x * 256 + threadIdx.x;
  if (i >= n) return;
  float v = in[i];
  if (act) v = silu_f(v);
  if (outf) outf[i] = v;
  if (outbf) outbf[i] = (bf16)v;
}

__global__ void zero_kernel(float* p, long n) {
  long i = (long)blockIdx.x * 256 + threadIdx.x;
  if (i < n) p[i] = 0.0f;
}

__global__ void add_kernel(const float* __restrict__ a, const float* __restrict__ b,
                           float* __restrict__ o, long n) {
  long i = (long)blockIdx.x * 256 + threadIdx.x;
  if (i < n) o[i] = a[i] + b[i];
}

// x_init = x + embd[bus_type]
__global__ void embed_kernel(const float* __restrict__ x, const int* __restrict__ bt,
                             const float* __restrict__ embd, float* __restrict__ xi,
                             long n16) {
  long i = (long)blockIdx.x * 256 + threadIdx.x;
  if (i >= n16) return;
  int n = (int)(i >> 4), c = (int)(i & 15);
  xi[i] = x[i] + embd[bt[n] * 16 + c];
}

// GroupNorm(C,C) with group size 1 is exactly: 0 * w + b  ->  broadcast gn_b
__global__ void fillgn_kernel(const float* __restrict__ gnb, bf16* __restrict__ out,
                              long n16) {
  long i = (long)blockIdx.x * 256 + threadIdx.x;
  if (i < n16) out[i] = (bf16)gnb[i & 15];
}

__global__ void deg_kernel(const int* __restrict__ ei, float* __restrict__ deg,
                           int E) {
  int e = blockIdx.x * 256 + threadIdx.x;
  if (e < E) atomicAdd(&deg[ei[E + e]], 1.0f);  // segment_sum(ones, col)
}

__global__ void norm_kernel(const int* __restrict__ ei, const float* __restrict__ deg,
                            float* __restrict__ nrm, int E) {
  int e = blockIdx.x * 256 + threadIdx.x;
  if (e >= E) return;
  float dr = deg[ei[e]], dc = deg[ei[E + e]];
  float a = dr > 0.0f ? rsqrtf(dr) : 0.0f;
  float b = dc > 0.0f ? rsqrtf(dc) : 0.0f;
  nrm[e] = a * b;
}

// hout[col] += norm * hin[row]   (TAGConv propagation, H=128)
__global__ void prop_kernel(const int* __restrict__ ei, const float* __restrict__ nrm,
                            const float* __restrict__ hin, float* __restrict__ hout,
                            int E) {
  long i = (long)blockIdx.x * 256 + threadIdx.x;
  if (i >= (long)E * 128) return;
  int e = (int)(i >> 7), f = (int)(i & 127);
  atomicAdd(&hout[(long)ei[E + e] * 128 + f], nrm[e] * hin[(long)ei[e] * 128 + f]);
}

// LayerNorm(128) + FiLM via precomputed [3,256] (s|t) table indexed by bus_type
__global__ __launch_bounds__(128) void ln_film_kernel(
    const float* __restrict__ x, const int* __restrict__ bt,
    const float* __restrict__ film, float* __restrict__ h, bf16* __restrict__ hbf,
    int N) {
  __shared__ float r1[128], r2[128];
  int n = blockIdx.x, j = threadIdx.x;
  float v = x[(long)n * 128 + j];
  r1[j] = v;
  r2[j] = v * v;
  __syncthreads();
  for (int s = 64; s > 0; s >>= 1) {
    if (j < s) { r1[j] += r1[j + s]; r2[j] += r2[j + s]; }
    __syncthreads();
  }
  float mean = r1[0] * (1.0f / 128.0f);
  float var = r2[0] * (1.0f / 128.0f) - mean * mean;
  float inv = rsqrtf(var + 1e-5f);
  const float* f = film + (long)bt[n] * 256;
  float hv = (v - mean) * inv * (1.0f + f[j]) + f[128 + j];
  h[(long)n * 128 + j] = hv;
  hbf[(long)n * 128 + j] = (bf16)hv;
}

// FiLM MLP for the 3 distinct embedding rows: 16 -> 128 (silu) -> 256
__global__ __launch_bounds__(256) void film_mlp_kernel(
    const float* __restrict__ embd, const float* __restrict__ w1,
    const float* __restrict__ b1, const float* __restrict__ w2,
    const float* __restrict__ b2, float* __restrict__ out) {
  __shared__ float hid[128];
  __shared__ float e[16];
  int j = threadIdx.x;
  for (int t = 0; t < 3; ++t) {
    if (j < 16) e[j] = embd[t * 16 + j];
    __syncthreads();
    if (j < 128) {
      float a = b1[j];
      for (int i = 0; i < 16; ++i) a += e[i] * w1[i * 128 + j];
      hid[j] = silu_f(a);
    }
    __syncthreads();
    float a = b2[j];
    for (int k = 0; k < 128; ++k) a += hid[k] * w2[k * 256 + j];
    out[t * 256 + j] = a;
    __syncthreads();
  }
}

// et layer 1: hidden = silu(ea @ W1 + b1), K=8 (too thin for WMMA) -> VALU
__global__ void et1_kernel(const float* __restrict__ ea, const float* __restrict__ w1,
                           const float* __restrict__ b1, bf16* __restrict__ out,
                           int E) {
  long i = (long)blockIdx.x * 256 + threadIdx.x;
  if (i >= (long)E * 128) return;
  int e = (int)(i >> 7), j = (int)(i & 127);
  const float* er = ea + (long)e * 8;
  float a = b1[j];
#pragma unroll
  for (int k = 0; k < 8; ++k) a += er[k] * w1[k * 128 + j];
  out[i] = (bf16)silu_f(a);
}

// Final grouped head: 4 x (32 -> 128, silu, 128 -> 4) + x_init
__global__ __launch_bounds__(256) void final_kernel(
    const float* __restrict__ x, const float* __restrict__ xi,
    const float* __restrict__ w1, const float* __restrict__ b1,
    const float* __restrict__ w2, const float* __restrict__ b2,
    float* __restrict__ out, int N) {
  __shared__ float z[512];
  int n = blockIdx.x, tid = threadIdx.x;
#pragma unroll
  for (int u = 0; u < 2; ++u) {
    int idx = tid + u * 256;
    int g = idx >> 7, j = idx & 127;
    const float* xr = x + (long)n * 128 + g * 32;
    float a = b1[g * 128 + j];
    for (int i = 0; i < 32; ++i) a += xr[i] * w1[(g * 32 + i) * 128 + j];
    z[idx] = silu_f(a);
  }
  __syncthreads();
  if (tid < 16) {
    int g = tid >> 2, o = tid & 3;
    float a = b2[g * 4 + o];
    for (int j = 0; j < 128; ++j) a += z[g * 128 + j] * w2[(g * 128 + j) * 4 + o];
    out[(long)n * 16 + tid] = a + xi[(long)n * 16 + tid];
  }
}

// ---------------------------------------------------------------------------
// Host orchestration
// Param leaves assumed in JAX sorted-key tree order after the 4 array inputs:
//  4 embd | 5-8 final{b1,b2,w1,w2} | 9 gn_b | 10 gn_w | 11-12 init_conv{b,w}
//  13-16 init_mp.ea{b1,b2,w1,w2} | 17-20 init_mp.et{b1,b2,w1,w2}
//  21+22*l mid[l]: bm1{b1,b2,w1,w2} bm2{...} conv{b,w} mlp{b1,b2,w1,w2}
//                  mp.ea{b1,b2,w1,w2} mp.et{b1,b2,w1,w2}
// ---------------------------------------------------------------------------
extern "C" void kernel_launch(void* const* d_in, const int* in_sizes, int n_in,
                              void* d_out, int out_size, void* d_ws, size_t ws_size,
                              hipStream_t stream) {
  (void)n_in; (void)out_size; (void)ws_size;
  const int NN = in_sizes[0] / 16;  // 20000 nodes
  const int E = in_sizes[2] / 2;    // 160000 edges
  const int H = 128;

  const float* x_in = (const float*)d_in[0];
  const int* bt = (const int*)d_in[1];
  const int* ei = (const int*)d_in[2];
  const float* ea = (const float*)d_in[3];
  const float* embd = (const float*)d_in[4];
  const float* fb1 = (const float*)d_in[5];
  const float* fb2 = (const float*)d_in[6];
  const float* fw1 = (const float*)d_in[7];
  const float* fw2 = (const float*)d_in[8];
  const float* gnb = (const float*)d_in[9];
  // d_in[10] = gn_w : unused — GroupNorm(C,C) output is exactly gn_b.
  const float* icb = (const float*)d_in[11];
  const float* icw = (const float*)d_in[12];
  const float* iea_b1 = (const float*)d_in[13];
  const float* iea_b2 = (const float*)d_in[14];
  const float* iea_w1 = (const float*)d_in[15];
  const float* iea_w2 = (const float*)d_in[16];
  const float* iet_b1 = (const float*)d_in[17];
  const float* iet_b2 = (const float*)d_in[18];
  const float* iet_w1 = (const float*)d_in[19];
  const float* iet_w2 = (const float*)d_in[20];
  const int MID0 = 21, MSTR = 22;
  auto MP = [&](int l, int k) { return (const float*)d_in[MID0 + l * MSTR + k]; };

  // ---- workspace carve ----
  char* ws = (char*)d_ws;
  size_t off = 0;
  auto carve = [&](size_t bytes) -> void* {
    void* p = ws + off;
    off = (off + bytes + 255) & ~(size_t)255;
    return p;
  };
  float* xinit = (float*)carve((size_t)NN * 16 * 4);
  bf16* x16bf = (bf16*)carve((size_t)NN * 16 * 2);
  float* film = (float*)carve((size_t)4 * 3 * 256 * 4);
  float* deg = (float*)carve((size_t)NN * 4);
  float* nrm = (float*)carve((size_t)E * 4);
  bf16* ebf = (bf16*)carve((size_t)E * 128 * 2);
  bf16* ethid = (bf16*)carve((size_t)E * 128 * 2);
  float* bufX = (float*)carve((size_t)NN * H * 4);
  float* bufH = (float*)carve((size_t)NN * H * 4);
  bf16* bufHbf = (bf16*)carve((size_t)NN * H * 2);
  float* mp = (float*)carve((size_t)NN * H * 4);
  bf16* mpbf = (bf16*)carve((size_t)NN * H * 2);
  float* tag = (float*)carve((size_t)NN * H * 4);
  float* p1 = (float*)carve((size_t)NN * H * 4);
  float* p2 = (float*)carve((size_t)NN * H * 4);
  bf16* pbf = (bf16*)carve((size_t)NN * H * 2);
  bf16* midhid = (bf16*)carve((size_t)NN * 512 * 2);
  // transposed bf16 weights [N][K]
  bf16* w_iea1 = (bf16*)carve((size_t)160 * 256 * 2);
  bf16* w_iea2 = (bf16*)carve((size_t)256 * 128 * 2);
  bf16* w_iet2 = (bf16*)carve((size_t)128 * 128 * 2);
  bf16* w_icw = (bf16*)carve((size_t)4 * 128 * 128 * 2);
  bf16 *w_ea1[2], *w_ea2[2], *w_et2[2], *w_cw[2], *w_m1[2], *w_m2[2];
  for (int l = 0; l < 2; ++l) {
    w_ea1[l] = (bf16*)carve((size_t)384 * 256 * 2);
    w_ea2[l] = (bf16*)carve((size_t)256 * 128 * 2);
    w_et2[l] = (bf16*)carve((size_t)128 * 128 * 2);
    w_cw[l] = (bf16*)carve((size_t)4 * 128 * 128 * 2);
    w_m1[l] = (bf16*)carve((size_t)128 * 512 * 2);
    w_m2[l] = (bf16*)carve((size_t)512 * 128 * 2);
  }

  auto cvt = [&](const float* s, bf16* d, long n) {
    convert_kernel<<<(unsigned)((n + 255) / 256), 256, 0, stream>>>(s, d, n);
  };
  auto cvt_tr = [&](const float* s, bf16* d, int K, int N) {
    long n = (long)K * N;
    convert_tr_kernel<<<(unsigned)((n + 255) / 256), 256, 0, stream>>>(s, d, K, N);
  };
  auto zero = [&](float* p, long n) {
    zero_kernel<<<(unsigned)((n + 255) / 256), 256, 0, stream>>>(p, n);
  };
  auto gemm = [&](const bf16* A, const bf16* Bt, const float* bias, const float* Cin,
                  float* Out, bf16* OutBf, int M, int N, int K, int act) {
    dim3 g((unsigned)(M / 32), (unsigned)(N / 128));
    if (act)
      gemm_kernel<1><<<g, 256, 0, stream>>>(A, Bt, bias, Cin, Out, OutBf, M, N, K);
    else
      gemm_kernel<0><<<g, 256, 0, stream>>>(A, Bt, bias, Cin, Out, OutBf, M, N, K);
  };

  // 0) weights -> transposed bf16 (once per launch; deterministic)
  cvt_tr(iea_w1, w_iea1, 160, 256);
  cvt_tr(iea_w2, w_iea2, 256, 128);
  cvt_tr(iet_w2, w_iet2, 128, 128);
  for (int k = 0; k < 4; ++k) cvt_tr(icw + k * H * H, w_icw + k * H * H, 128, 128);
  for (int l = 0; l < 2; ++l) {
    cvt_tr(MP(l, 16), w_ea1[l], 384, 256);
    cvt_tr(MP(l, 17), w_ea2[l], 256, 128);
    cvt_tr(MP(l, 21), w_et2[l], 128, 128);
    for (int k = 0; k < 4; ++k)
      cvt_tr(MP(l, 9) + k * H * H, w_cw[l] + k * H * H, 128, 128);
    cvt_tr(MP(l, 12), w_m1[l], 128, 512);
    cvt_tr(MP(l, 13), w_m2[l], 512, 128);
  }

  // 1) embedding residual + exact GroupNorm(C,C) result
  embed_kernel<<<(unsigned)((NN * 16 + 255) / 256), 256, 0, stream>>>(
      x_in, bt, embd, xinit, (long)NN * 16);
  fillgn_kernel<<<(unsigned)((NN * 16 + 255) / 256), 256, 0, stream>>>(
      gnb, x16bf, (long)NN * 16);

  // 2) TAGConv degree normalization (on original, un-doubled edges)
  zero(deg, NN);
  deg_kernel<<<(unsigned)((E + 255) / 256), 256, 0, stream>>>(ei, deg, E);
  norm_kernel<<<(unsigned)((E + 255) / 256), 256, 0, stream>>>(ei, deg, nrm, E);

  // 3) FiLM tables for the 3 bus types
  for (int l = 0; l < 2; ++l) {
    film_mlp_kernel<<<1, 256, 0, stream>>>(embd, MP(l, 2), MP(l, 0), MP(l, 3),
                                           MP(l, 1), film + (l * 2 + 0) * 768);
    film_mlp_kernel<<<1, 256, 0, stream>>>(embd, MP(l, 6), MP(l, 4), MP(l, 7),
                                           MP(l, 5), film + (l * 2 + 1) * 768);
  }

  auto do_et = [&](const float* w1f, const float* b1f, const bf16* w2t,
                   const float* b2f) {
    et1_kernel<<<(unsigned)(((long)E * 128 + 255) / 256), 256, 0, stream>>>(
        ea, w1f, b1f, ethid, E);
    gemm(ethid, w2t, b2f, nullptr, nullptr, ebf, E, H, H, 0);
  };

  auto tagconv = [&](const float* hf, const bf16* hb, const float* cb,
                     const bf16* cwt, float* xout) {
    gemm(hb, cwt, cb, nullptr, xout, nullptr, NN, H, H, 0);  // x @ W0 + b
    const float* cur = hf;
    float* pa = p1;
    float* pb = p2;
    for (int k = 1; k <= 3; ++k) {
      zero(pa, (long)NN * H);
      prop_kernel<<<(unsigned)(((long)E * 128 + 255) / 256), 256, 0, stream>>>(
          ei, nrm, cur, pa, E);
      cvt(pa, pbf, (long)NN * H);
      gemm(pbf, cwt + (size_t)k * H * H, nullptr, xout, xout, nullptr, NN, H, H, 0);
      cur = pa;
      float* t = pa; pa = pb; pb = t;
    }
  };

  // 4) init: et + edge message passing (NF=16) + TAGConv
  do_et(iet_w1, iet_b1, w_iet2, iet_b2);
  zero(mp, (long)NN * H);
  edge_mp_kernel<16><<<(unsigned)(2 * E / 32), 256, 0, stream>>>(
      ei, E, x16bf, ebf, w_iea1, iea_b1, w_iea2, iea_b2, mp);
  ew_kernel<<<(unsigned)(((long)NN * H + 255) / 256), 256, 0, stream>>>(
      mp, nullptr, mpbf, (long)NN * H, 0);
  tagconv(mp, mpbf, icb, w_icw, bufX);

  // 5) mid layers
  for (int l = 0; l < 2; ++l) {
    // block 1: LN + FiLM -> edge MP -> silu -> TAGConv -> residual
    ln_film_kernel<<<(unsigned)NN, 128, 0, stream>>>(
        bufX, bt, film + (l * 2 + 0) * 768, bufH, bufHbf, NN);
    do_et(MP(l, 20), MP(l, 18), w_et2[l], MP(l, 19));
    zero(mp, (long)NN * H);
    edge_mp_kernel<128><<<(unsigned)(2 * E / 32), 256, 0, stream>>>(
        ei, E, bufHbf, ebf, w_ea1[l], MP(l, 14), w_ea2[l], MP(l, 15), mp);
    ew_kernel<<<(unsigned)(((long)NN * H + 255) / 256), 256, 0, stream>>>(
        mp, mp, mpbf, (long)NN * H, 1);
    tagconv(mp, mpbf, MP(l, 8), w_cw[l], tag);
    add_kernel<<<(unsigned)(((long)NN * H + 255) / 256), 256, 0, stream>>>(
        tag, bufX, bufX, (long)NN * H);
    // block 2: LN + FiLM -> MLP(128->512->128) -> residual
    ln_film_kernel<<<(unsigned)NN, 128, 0, stream>>>(
        bufX, bt, film + (l * 2 + 1) * 768, bufH, bufHbf, NN);
    gemm(bufHbf, w_m1[l], MP(l, 10), nullptr, nullptr, midhid, NN, 512, H, 1);
    gemm(midhid, w_m2[l], MP(l, 11), bufX, bufX, nullptr, NN, H, 512, 0);
  }

  // 6) final grouped head + x_init residual
  final_kernel<<<(unsigned)NN, 256, 0, stream>>>(bufX, xinit, fw1, fb1, fw2, fb2,
                                                 (float*)d_out, NN);
}
